// SwitchFeedForward_32993938768510
// MI455X (gfx1250) — compile-verified, matrix operand order checked by reading, not run on hardware
//
#include <hip/hip_runtime.h>
#include <math.h>

// Problem constants (match reference)
#define T_TOK 8192
#define D_HID 1024
#define FF_DIM 4096
#define N_EXP 8
#define CAP 1024

typedef __attribute__((ext_vector_type(16))) _Float16 v16h;
typedef __attribute__((ext_vector_type(8)))  float    v8f;

// GEMM tiling: 128x128 output per workgroup, K-step 32.
// 8 waves (wave32) arranged 2x4; each wave computes 64x32 via 4x2 WMMA 16x16 tiles.
#define BM 128
#define BN 128
#define BK 32
#define BSTR 40   // LDS row stride (halfs) for transposed B tile: [n][k]

// Workspace layout (bytes). Needs ~80 MB + 1 MB header.
#define WS_ROUTES 0
#define WS_FACTOR (T_TOK * 4)
#define WS_S2T    (T_TOK * 8)
#define WS_BUF    ((size_t)1 << 20)
#define WS_H      (WS_BUF + (size_t)N_EXP * CAP * D_HID * 2)

// CDNA5 async global->LDS copies (ASYNCcnt), guarded so compile never regresses.
// Probe round 2: param1 is 'int __vector(4) __device__*'  => v4i in AS1 (global),
// param2 expected to be the LDS-side pointer (AS3).
#if __has_builtin(__builtin_amdgcn_global_load_async_to_lds_b128) && \
    __has_builtin(__builtin_amdgcn_s_wait_asynccnt)
#define USE_ASYNC 1
typedef int v4i __attribute__((vector_size(16)));
typedef __attribute__((address_space(1))) v4i as1_v4i;
typedef __attribute__((address_space(3))) v4i as3_v4i;
#else
#define USE_ASYNC 0
#endif

union AFrag { v16h h; unsigned u[8]; };

// ---------------- small kernels ----------------

__global__ void k_init_s2t(int* __restrict__ s2t) {
    int i = blockIdx.x * blockDim.x + threadIdx.x;
    if (i < N_EXP * CAP) s2t[i] = -1;
}

// One wave32 per token: logits = x_row @ W_sw + b_sw; softmax-max; argmax (first max).
__global__ void k_router(const float* __restrict__ x, const float* __restrict__ Wsw,
                         const float* __restrict__ bsw,
                         int* __restrict__ routes, float* __restrict__ factor) {
    int wave = threadIdx.x >> 5;
    int lane = threadIdx.x & 31;
    int t = blockIdx.x * 8 + wave;
    if (t >= T_TOK) return;
    const float* xr = x + (size_t)t * D_HID;
    float acc[N_EXP];
#pragma unroll
    for (int e = 0; e < N_EXP; ++e) acc[e] = 0.f;
    for (int i = 0; i < D_HID / 32; ++i) {
        int d = lane + i * 32;
        float xv = xr[d];
        const float4* w = (const float4*)(Wsw + (size_t)d * N_EXP);
        float4 w0 = w[0], w1 = w[1];
        acc[0] += xv * w0.x; acc[1] += xv * w0.y;
        acc[2] += xv * w0.z; acc[3] += xv * w0.w;
        acc[4] += xv * w1.x; acc[5] += xv * w1.y;
        acc[6] += xv * w1.z; acc[7] += xv * w1.w;
    }
#pragma unroll
    for (int e = 0; e < N_EXP; ++e)
#pragma unroll
        for (int off = 16; off > 0; off >>= 1)
            acc[e] += __shfl_xor(acc[e], off, 32);
    if (lane == 0) {
        float l[N_EXP];
        float lmax = -1e30f; int amax = 0;
#pragma unroll
        for (int e = 0; e < N_EXP; ++e) {
            l[e] = acc[e] + bsw[e];
            if (l[e] > lmax) { lmax = l[e]; amax = e; }
        }
        float s = 0.f;
#pragma unroll
        for (int e = 0; e < N_EXP; ++e) s += expf(l[e] - lmax);
        float pmax = 1.0f / s;          // = exp(0)/sum
        routes[t] = amax;
        factor[t] = pmax / pmax;        // forward value == 1.0
    }
}

// Sequential-order capacity assignment: ballot prefix over 32 tokens at a time.
__global__ void k_scan(const int* __restrict__ routes, int* __restrict__ s2t) {
    int lane = threadIdx.x;
    unsigned lt = (1u << lane) - 1u;
    int base[N_EXP];
#pragma unroll
    for (int e = 0; e < N_EXP; ++e) base[e] = 0;
    for (int t0 = 0; t0 < T_TOK; t0 += 32) {
        int r = routes[t0 + lane];
#pragma unroll
        for (int e = 0; e < N_EXP; ++e) {
            unsigned m = (unsigned)__ballot(r == e);
            if (r == e) {
                int p = base[e] + __popc(m & lt);
                if (p < CAP) s2t[(size_t)e * CAP + p] = t0 + lane;
            }
            base[e] += __popc(m);
        }
    }
}

// rows [0, E*CAP): fill f16 expert buffer (zeros for empty slots)
// rows [E*CAP, E*CAP+T): write default output final = x * factor (dropped-token path)
__global__ void k_scatter(const float* __restrict__ x, const float* __restrict__ factor,
                          const int* __restrict__ s2t, _Float16* __restrict__ buf,
                          float* __restrict__ outp) {
    int row = blockIdx.x;
    int c = threadIdx.x * 4;
    if (row < N_EXP * CAP) {
        int t = s2t[row];
        union { _Float16 h[4]; uint2 u; } v;
        if (t >= 0) {
            float f = factor[t];
            float4 xv = *(const float4*)(x + (size_t)t * D_HID + c);
            v.h[0] = (_Float16)(xv.x * f); v.h[1] = (_Float16)(xv.y * f);
            v.h[2] = (_Float16)(xv.z * f); v.h[3] = (_Float16)(xv.w * f);
        } else {
            v.h[0] = v.h[1] = v.h[2] = v.h[3] = (_Float16)0.f;
        }
        *(uint2*)(buf + (size_t)row * D_HID + c) = v.u;
    } else {
        int t = row - N_EXP * CAP;
        float f = factor[t];
        float4 xv = *(const float4*)(x + (size_t)t * D_HID + c);
        xv.x *= f; xv.y *= f; xv.z *= f; xv.w *= f;
        *(float4*)(outp + (size_t)t * D_HID + c) = xv;
    }
}

// ---------------- GEMM helpers ----------------

__device__ __forceinline__ void async_wait() {
#if USE_ASYNC
    __builtin_amdgcn_s_wait_asynccnt(0);
#endif
}

// A tile: 128 rows x 32 halfs of f16 -> LDS, pure byte copy.
// Async path: 2x global_load_async_to_lds_b128 per thread (ASYNCcnt-tracked).
__device__ __forceinline__ void issue_a_tile(_Float16* As, const _Float16* A,
                                             int lda, int m0, int k0, int tid) {
#pragma unroll
    for (int it = 0; it < 2; ++it) {
        int idx = tid + it * 256;       // 512 x 16B
        int r = idx >> 2;
        int c4 = idx & 3;
        const _Float16* g = A + (size_t)(m0 + r) * lda + k0 + c4 * 8;
        _Float16* l = As + r * BK + c4 * 8;
#if USE_ASYNC
        __builtin_amdgcn_global_load_async_to_lds_b128(
            (as1_v4i*)(unsigned long long)(uintptr_t)g,
            (as3_v4i*)(unsigned)(uintptr_t)l, 0, 0);
#else
        *(uint4*)l = *(const uint4*)g;
#endif
    }
}

// B tile: 32 k-rows x 128 n-cols f32. Stage 1: global float4 loads into regs
// (paired k rows). Stage 2: convert and store transposed [n][k] with packed
// (k,k+1) dword DS stores.
struct BReg { float4 a, b; };

__device__ __forceinline__ void load_b_regs(BReg br[2], const float* B,
                                            int ldb, int k0, int n0, int tid) {
#pragma unroll
    for (int it = 0; it < 2; ++it) {
        int idx = tid + it * 256;       // 0..511
        int kp = idx >> 5;              // k pair 0..15
        int c4 = (idx & 31) * 4;        // col 0..124
        const float* p = B + (size_t)(k0 + 2 * kp) * ldb + n0 + c4;
        br[it].a = *(const float4*)p;
        br[it].b = *(const float4*)(p + ldb);
    }
}

__device__ __forceinline__ void store_b_lds(_Float16* Bs, const BReg br[2], int tid) {
#pragma unroll
    for (int it = 0; it < 2; ++it) {
        int idx = tid + it * 256;
        int kp = idx >> 5;
        int c4 = (idx & 31) * 4;
        const float* lo = &br[it].a.x;
        const float* hi = &br[it].b.x;
#pragma unroll
        for (int i = 0; i < 4; ++i) {
            union { _Float16 h[2]; unsigned u; } p;
            p.h[0] = (_Float16)lo[i];   // k = 2*kp
            p.h[1] = (_Float16)hi[i];   // k = 2*kp+1
            *(unsigned*)(Bs + (c4 + i) * BSTR + 2 * kp) = p.u;
        }
    }
}

// ISA 16-bit A layout (16x32): lane L holds row L%16; VGPR j holds K pair
// kb = (j<4 ? 2j : 16+2(j-4)) + 8*(L>=16).
__device__ __forceinline__ v16h load_a_frag(const _Float16* As, int mbase, int lane) {
    AFrag f;
    int row = mbase + (lane & 15);
    int kofs = (lane >> 4) * 8;
#pragma unroll
    for (int j = 0; j < 8; ++j) {
        int kb = (j < 4 ? 2 * j : 16 + 2 * (j - 4)) + kofs;
        f.u[j] = *(const unsigned*)(As + row * BK + kb);
    }
    return f.h;
}

// B operand mirrors A with N in place of M (B stored transposed [n][k] in LDS).
__device__ __forceinline__ v16h load_b_frag(const _Float16* Bs, int nbase, int lane) {
    AFrag f;
    int col = nbase + (lane & 15);
    int kofs = (lane >> 4) * 8;
#pragma unroll
    for (int j = 0; j < 8; ++j) {
        int kb = (j < 4 ? 2 * j : 16 + 2 * (j - 4)) + kofs;
        f.u[j] = *(const unsigned*)(Bs + col * BSTR + kb);
    }
    return f.h;
}

// Double-buffered main loop: async A copies + register-staged B for tile k+1
// overlap the WMMAs of tile k. One own-wave async wait + barrier per iteration.
__device__ __forceinline__ void gemm_loop(v8f acc[4][2],
                                          const _Float16* A, int lda,
                                          const float* B, int ldb, int K,
                                          int m0, int n0, int wm, int wn,
                                          int tid, int lane,
                                          _Float16* As, _Float16* Bs) {
    BReg br[2];
    issue_a_tile(As, A, lda, m0, 0, tid);
    load_b_regs(br, B, ldb, 0, n0, tid);
    store_b_lds(Bs, br, tid);
    async_wait();
    __syncthreads();

    int nk = K / BK;
    for (int kt = 0; kt < nk; ++kt) {
        int cur = kt & 1, nxt = cur ^ 1;
        const _Float16* Asc = As + cur * (BM * BK);
        const _Float16* Bsc = Bs + cur * (BN * BSTR);
        if (kt + 1 < nk) {
            issue_a_tile(As + nxt * (BM * BK), A, lda, m0, (kt + 1) * BK, tid);
            load_b_regs(br, B, ldb, (kt + 1) * BK, n0, tid);
        }
        v16h af[4], bf[2];
#pragma unroll
        for (int i = 0; i < 4; ++i) af[i] = load_a_frag(Asc, wm + i * 16, lane);
#pragma unroll
        for (int j = 0; j < 2; ++j) bf[j] = load_b_frag(Bsc, wn + j * 16, lane);
#pragma unroll
        for (int i = 0; i < 4; ++i)
#pragma unroll
            for (int j = 0; j < 2; ++j)
                acc[i][j] = __builtin_amdgcn_wmma_f32_16x16x32_f16(
                    false, af[i], false, bf[j], (short)0, acc[i][j], false, false);
        if (kt + 1 < nk) store_b_lds(Bs + nxt * (BN * BSTR), br, tid);
        async_wait();
        __syncthreads();
    }
}

// ---------------- GEMM1: h = gelu(buf_e @ W1_e + b1_e), f16 out ----------------

__global__ void __launch_bounds__(256) k_gemm1(const _Float16* __restrict__ buf,
                                               const float* __restrict__ W1,
                                               const float* __restrict__ b1,
                                               _Float16* __restrict__ h) {
    __shared__ __align__(16) _Float16 As[2 * BM * BK];
    __shared__ __align__(16) _Float16 Bs[2 * BN * BSTR];
    int e = blockIdx.z;
    int m0 = blockIdx.y * BM;
    int n0 = blockIdx.x * BN;
    int tid = threadIdx.x;
    int wave = tid >> 5, lane = tid & 31;
    int wm = (wave >> 2) * 64;
    int wn = (wave & 3) * 32;
    const _Float16* A = buf + (size_t)e * CAP * D_HID;
    const float* B = W1 + (size_t)e * D_HID * FF_DIM;

    v8f acc[4][2];
#pragma unroll
    for (int i = 0; i < 4; ++i)
#pragma unroll
        for (int j = 0; j < 2; ++j)
#pragma unroll
            for (int q = 0; q < 8; ++q) acc[i][j][q] = 0.f;

    gemm_loop(acc, A, D_HID, B, FF_DIM, D_HID, m0, n0, wm, wn, tid, lane, As, Bs);

    // C/D layout: lane L -> col L%16; VGPR r -> row r + 8*(L>=16)
    int halfo = (lane >> 4) * 8;
    int colL = lane & 15;
#pragma unroll
    for (int i = 0; i < 4; ++i) {
#pragma unroll
        for (int j = 0; j < 2; ++j) {
            int colg = n0 + wn + j * 16 + colL;
            float bias = b1[(size_t)e * FF_DIM + colg];
#pragma unroll
            for (int r = 0; r < 8; ++r) {
                int rowg = m0 + wm + i * 16 + r + halfo;
                float v = acc[i][j][r] + bias;
                float g = v * 0.5f * (1.0f + erff(v * (1.0f / 1.41421f)));
                h[((size_t)e * CAP + rowg) * FF_DIM + colg] = (_Float16)g;
            }
        }
    }
}

// ------- GEMM2: out[token] = h_e @ W2_e + b2_e, scattered via slot->token -------

__global__ void __launch_bounds__(256) k_gemm2(const _Float16* __restrict__ h,
                                               const float* __restrict__ W2,
                                               const float* __restrict__ b2,
                                               const int* __restrict__ s2t,
                                               float* __restrict__ outp) {
    __shared__ __align__(16) _Float16 As[2 * BM * BK];
    __shared__ __align__(16) _Float16 Bs[2 * BN * BSTR];
    int e = blockIdx.z;
    int m0 = blockIdx.y * BM;
    int n0 = blockIdx.x * BN;
    int tid = threadIdx.x;
    int wave = tid >> 5, lane = tid & 31;
    int wm = (wave >> 2) * 64;
    int wn = (wave & 3) * 32;
    const _Float16* A = h + (size_t)e * CAP * FF_DIM;
    const float* B = W2 + (size_t)e * FF_DIM * D_HID;

    v8f acc[4][2];
#pragma unroll
    for (int i = 0; i < 4; ++i)
#pragma unroll
        for (int j = 0; j < 2; ++j)
#pragma unroll
            for (int q = 0; q < 8; ++q) acc[i][j][q] = 0.f;

    gemm_loop(acc, A, FF_DIM, B, D_HID, FF_DIM, m0, n0, wm, wn, tid, lane, As, Bs);

    int halfo = (lane >> 4) * 8;
    int colL = lane & 15;
#pragma unroll
    for (int i = 0; i < 4; ++i) {
#pragma unroll
        for (int r = 0; r < 8; ++r) {
            int rowg = m0 + wm + i * 16 + r + halfo;   // slot within expert
            int t = s2t[(size_t)e * CAP + rowg];
            if (t < 0) continue;                       // unfilled slot: keep default xs
#pragma unroll
            for (int j = 0; j < 2; ++j) {
                int colg = n0 + wn + j * 16 + colL;
                outp[(size_t)t * D_HID + colg] = acc[i][j][r] + b2[(size_t)e * D_HID + colg];
            }
        }
    }
}

// ---------------- launch ----------------

extern "C" void kernel_launch(void* const* d_in, const int* in_sizes, int n_in,
                              void* d_out, int out_size, void* d_ws, size_t ws_size,
                              hipStream_t stream) {
    const float* x   = (const float*)d_in[0];
    const float* Wsw = (const float*)d_in[1];
    const float* bsw = (const float*)d_in[2];
    const float* W1  = (const float*)d_in[3];
    const float* b1  = (const float*)d_in[4];
    const float* W2  = (const float*)d_in[5];
    const float* b2  = (const float*)d_in[6];
    float* outp = (float*)d_out;

    char* ws = (char*)d_ws;
    int*      routes = (int*)(ws + WS_ROUTES);
    float*    factor = (float*)(ws + WS_FACTOR);
    int*      s2t    = (int*)(ws + WS_S2T);
    _Float16* buf    = (_Float16*)(ws + WS_BUF);   // E*CAP x D f16 (16 MB)
    _Float16* hbuf   = (_Float16*)(ws + WS_H);     // E*CAP x FF f16 (64 MB)

    k_init_s2t<<<(N_EXP * CAP + 255) / 256, 256, 0, stream>>>(s2t);
    k_router<<<T_TOK / 8, 256, 0, stream>>>(x, Wsw, bsw, routes, factor);
    k_scan<<<1, 32, 0, stream>>>(routes, s2t);
    k_scatter<<<N_EXP * CAP + T_TOK, 256, 0, stream>>>(x, factor, s2t, buf, outp);
    k_gemm1<<<dim3(FF_DIM / BN, CAP / BM, N_EXP), 256, 0, stream>>>(buf, W1, b1, hbuf);
    k_gemm2<<<dim3(D_HID / BN, CAP / BM, N_EXP), 256, 0, stream>>>(hbuf, W2, b2, s2t, outp);
}
